// MambaLM_18829136626171
// MI455X (gfx1250) — matrix operand order repaired and checked by reading, not compiled, not run on hardware
//
#include <hip/hip_runtime.h>
#include <hip/hip_bf16.h>
#include <stdint.h>

// ---- model constants -------------------------------------------------------
static constexpr int cV  = 32000, cD = 1024, cNL = 4;
static constexpr int cE  = 2048,  cS = 16,  cR  = 64, cK = 4;
static constexpr int cB  = 2,     cL = 1024;
static constexpr int cM  = cB * cL;          // 2048 token rows
static constexpr int cDBC = cR + 2 * cS;     // 96

// ---- WMMA vector types -----------------------------------------------------
typedef __attribute__((ext_vector_type(16))) __bf16          v16bf;
typedef __attribute__((ext_vector_type(16))) unsigned short  v16us;
typedef __attribute__((ext_vector_type(8)))  float           v8f;

__device__ __forceinline__ unsigned short f2bf(float f) {
  unsigned int u = __builtin_bit_cast(unsigned int, f);
  u += 0x7FFFu + ((u >> 16) & 1u);           // round-to-nearest-even
  return (unsigned short)(u >> 16);
}

__device__ __forceinline__ unsigned int pack_bf2(float f0, float f1) {
  return (unsigned int)f2bf(f0) | ((unsigned int)f2bf(f1) << 16);
}

__device__ __forceinline__ float siluf(float x) {
  return x / (1.0f + __expf(-x));
}

// ---- generic bf16-WMMA GEMM: Out[M,N] = A[M,K]*Bw[K,N] (+ Cin) -------------
// A row-major (leading dim lda), Bw row-major (K x N), f32 in/out.
// LDS tiles are stored in *fragment order* so each lane reads its whole
// 16xbf16 WMMA fragment with one 32B vector load (2x ds_load_b128).
// Each wave owns TWO 16-row M-tiles (tiles wave and wave+4) so every B
// fragment read feeds 2 WMMAs: 16 WMMA per wave per staging round.
//
// Fragment maps (ISA 7.12.2, wave32):
//   A (16x32 bf16): lane = m + 16*half;  k -> half=(k>>3)&1, e=(k&7)+(k&16?8:0)
//   B (32x16 bf16): lane = n + 16*half;  k -> half=(k>>3)&1, e=(k&7)+(k&16?8:0)
//   C/D (16x16 f32): m = vgpr + 8*half,  n = lane&15
#define BM 128
#define BN 64
#define BK 64           // two WMMA k-steps per staging round
__global__ __launch_bounds__(128) void gemm_bf16_wmma(
    const float* __restrict__ A, int lda,
    const float* __restrict__ Bw,
    const float* __restrict__ Cin,      // optional residual (may be null)
    float* __restrict__ Out,
    int M, int N, int Kd)               // Kd % 64 == 0
{
  // sA: [tile(8)][ks(2)][lane(32)][elem(16)] u16 = 16 KB
  // sB: [tile(4)][ks(2)][lane(32)][elem(16)] u16 =  8 KB
  __shared__ __align__(32) unsigned short sA[8 * 2 * 32 * 16];
  __shared__ __align__(32) unsigned short sB[4 * 2 * 32 * 16];
  unsigned int* sA32 = (unsigned int*)sA;
  unsigned int* sB32 = (unsigned int*)sB;

  const int bm   = blockIdx.y * BM;
  const int bn   = blockIdx.x * BN;
  const int tid  = threadIdx.x;            // 0..127 (4 wave32)
  const int wave = tid >> 5;
  const int lane = tid & 31;
  const int half = lane >> 4;
  const int mn   = lane & 15;

  v8f acc[2][4];
  #pragma unroll
  for (int p = 0; p < 2; ++p)
    #pragma unroll
    for (int i = 0; i < 4; ++i) acc[p][i] = (v8f){0,0,0,0,0,0,0,0};

  for (int k0 = 0; k0 < Kd; k0 += BK) {
    // ---- stage A tile (BM x BK) into fragment order, paired bf16 stores ----
    #pragma unroll
    for (int i = tid; i < BM * (BK / 2); i += 128) {
      int m_local = i >> 5;                 // 0..127, 32 k-pairs per row
      int k       = (i & 31) * 2;
      int gr      = bm + m_local;
      float f0 = 0.f, f1 = 0.f;
      if (gr < M) {
        const float* ap = A + (size_t)gr * lda + k0 + k;
        f0 = ap[0]; f1 = ap[1];
      }
      int ks  = k >> 5, kk = k & 31;
      int haf = (kk >> 3) & 1;
      int e   = (kk & 7) + ((kk & 16) ? 8 : 0);       // even
      int t   = m_local >> 4, m = m_local & 15;       // t in 0..7
      sA32[(t * 1024 + ks * 512 + (m + 16 * haf) * 16 + e) >> 1] = pack_bf2(f0, f1);
    }
    // ---- stage B tile (BK x BN) into fragment order ------------------------
    #pragma unroll
    for (int i = tid; i < (BK / 2) * BN; i += 128) {
      int k       = (i >> 6) * 2;           // 0..62
      int n_local = i & 63;
      int gc      = bn + n_local;
      float f0 = 0.f, f1 = 0.f;
      if (gc < N) {
        f0 = Bw[(size_t)(k0 + k) * N + gc];
        f1 = Bw[(size_t)(k0 + k + 1) * N + gc];
      }
      int ks  = k >> 5, kk = k & 31;
      int haf = (kk >> 3) & 1;
      int e   = (kk & 7) + ((kk & 16) ? 8 : 0);       // even
      int nt  = n_local >> 4, n = n_local & 15;
      sB32[(nt * 1024 + ks * 512 + (n + 16 * haf) * 16 + e) >> 1] = pack_bf2(f0, f1);
    }
    __syncthreads();

    // ---- 16 WMMAs per wave: each B fragment feeds 2 WMMAs ------------------
    #pragma unroll
    for (int ks = 0; ks < 2; ++ks) {
      v16us au0 = *((const v16us*)&sA[(wave    ) * 1024 + ks * 512 + lane * 16]);
      v16us au1 = *((const v16us*)&sA[(wave + 4) * 1024 + ks * 512 + lane * 16]);
      v16bf a0 = __builtin_bit_cast(v16bf, au0);
      v16bf a1 = __builtin_bit_cast(v16bf, au1);
      #pragma unroll
      for (int nt = 0; nt < 4; ++nt) {
        v16us bu = *((const v16us*)&sB[nt * 1024 + ks * 512 + lane * 16]);
        v16bf bfrag = __builtin_bit_cast(v16bf, bu);
        acc[0][nt] = __builtin_amdgcn_wmma_f32_16x16x32_bf16(
            false, a0, false, bfrag, (short)0, acc[0][nt], false, false);
        acc[1][nt] = __builtin_amdgcn_wmma_f32_16x16x32_bf16(
            false, a1, false, bfrag, (short)0, acc[1][nt], false, false);
      }
    }
    __syncthreads();
  }

  // ---- write D: m = r + half*8 (+tile*16), n = mn (+nt*16) -----------------
  #pragma unroll
  for (int p = 0; p < 2; ++p) {
    #pragma unroll
    for (int nt = 0; nt < 4; ++nt) {
      #pragma unroll
      for (int r = 0; r < 8; ++r) {
        int gm = bm + (wave + p * 4) * 16 + half * 8 + r;
        int gn = bn + nt * 16 + mn;
        if (gm < M && gn < N) {
          float v = acc[p][nt][r];
          if (Cin) v += Cin[(size_t)gm * N + gn];
          Out[(size_t)gm * N + gn] = v;
        }
      }
    }
  }
}

// ---- embedding gather ------------------------------------------------------
__global__ __launch_bounds__(256) void embed_kernel(
    const int* __restrict__ tok, const float* __restrict__ mask,
    const float* __restrict__ emb, float* __restrict__ h)
{
  int i = blockIdx.x * 256 + threadIdx.x;
  if (i >= cM * cD) return;
  int row = i / cD, d = i - row * cD;
  h[i] = emb[(size_t)tok[row] * cD + d] * mask[row];
}

// ---- block reduction helper ------------------------------------------------
__device__ __forceinline__ float block_reduce_sum256(float v, float* sm) {
  int tid = threadIdx.x;
  sm[tid] = v; __syncthreads();
  #pragma unroll
  for (int s = 128; s > 0; s >>= 1) {
    if (tid < s) sm[tid] += sm[tid + s];
    __syncthreads();
  }
  float r = sm[0]; __syncthreads();
  return r;
}

// ---- RMSNorm ---------------------------------------------------------------
__global__ __launch_bounds__(256) void rmsnorm_kernel(
    const float* __restrict__ x, const float* __restrict__ w,
    float* __restrict__ out)
{
  __shared__ float sm[256];
  int row = blockIdx.x;
  const float* xr = x + (size_t)row * cD;
  float s = 0.f;
  for (int d = threadIdx.x; d < cD; d += 256) { float v = xr[d]; s += v * v; }
  float tot = block_reduce_sum256(s, sm);
  float scale = rsqrtf(tot / (float)cD + 1e-5f);
  for (int d = threadIdx.x; d < cD; d += 256)
    out[(size_t)row * cD + d] = xr[d] * scale * w[d];
}

// ---- causal depthwise conv (K=4) + bias + SiLU -----------------------------
__global__ __launch_bounds__(256) void conv_silu_kernel(
    const float* __restrict__ uz,          // (cM, 2E), u in first E cols
    const float* __restrict__ cw,          // (E, K)
    const float* __restrict__ cb,          // (E)
    float* __restrict__ u)                 // (cM, E)
{
  int i = blockIdx.x * 256 + threadIdx.x;
  if (i >= cM * cE) return;
  int e  = i % cE;
  int bl = i / cE;
  int l  = bl % cL;
  int b  = bl / cL;
  float acc = cb[e];
  #pragma unroll
  for (int k = 0; k < cK; ++k) {
    int ls = l - (cK - 1) + k;
    if (ls >= 0)
      acc += uz[((size_t)(b * cL + ls)) * (2 * cE) + e] * cw[e * cK + k];
  }
  u[i] = siluf(acc);
}

// ---- delta = softplus(pre + b_dt[e]) (in place) ----------------------------
__global__ __launch_bounds__(256) void softplus_kernel(
    float* __restrict__ delta, const float* __restrict__ b_dt)
{
  int i = blockIdx.x * 256 + threadIdx.x;
  if (i >= cM * cE) return;
  int e = i % cE;
  float t = delta[i] + b_dt[e];
  delta[i] = fmaxf(t, 0.f) + log1pf(__expf(-fabsf(t)));
}

// ---- selective scan: lane owns one (b,e) chain, 16 states in registers -----
__global__ __launch_bounds__(256) void scan_kernel(
    const float* __restrict__ u,        // (cM, E)
    const float* __restrict__ delta,    // (cM, E)
    const float* __restrict__ A_log,    // (E, S)
    const float* __restrict__ dbc,      // (cM, 96): [dr|Bm|Cm]
    const float* __restrict__ Dsk,      // (E)
    float* __restrict__ y)              // (cM, E)
{
  int e = blockIdx.x * 256 + threadIdx.x;
  int b = blockIdx.y;
  if (e >= cE) return;

  float Acoef[cS];
  #pragma unroll
  for (int s = 0; s < cS; ++s) Acoef[s] = -__expf(A_log[(size_t)e * cS + s]);
  float hst[cS];
  #pragma unroll
  for (int s = 0; s < cS; ++s) hst[s] = 0.f;
  float dsk = Dsk[e];

  for (int l = 0; l < cL; ++l) {
    size_t row = (size_t)(b * cL + l);
    float dl = delta[row * cE + e];
    float ul = u[row * cE + e];
    float du = dl * ul;
    const float* bc = dbc + row * cDBC;   // uniform across block -> scalar
    float yv = 0.f;
    #pragma unroll
    for (int s = 0; s < cS; ++s) {
      float dA = __expf(dl * Acoef[s]);
      hst[s] = dA * hst[s] + du * bc[cR + s];
      yv += hst[s] * bc[cR + cS + s];
    }
    y[row * cE + e] = yv + ul * dsk;
  }
}

// ---- y *= silu(z) (in place) -----------------------------------------------
__global__ __launch_bounds__(256) void gate_kernel(
    float* __restrict__ y, const float* __restrict__ uz)
{
  int i = blockIdx.x * 256 + threadIdx.x;
  if (i >= cM * cE) return;
  int row = i / cE, e = i - row * cE;
  float z = uz[(size_t)row * (2 * cE) + cE + e];
  y[i] *= siluf(z);
}

// ---- wbar[d] = mean_s W_rw[d,s]; wbar[D] = mean(b_rw) ----------------------
__global__ __launch_bounds__(256) void wbar_kernel(
    const float* __restrict__ W_rw, const float* __restrict__ b_rw,
    float* __restrict__ wbar)
{
  for (int d = threadIdx.x; d < cD; d += 256) {
    float s = 0.f;
    #pragma unroll
    for (int k = 0; k < cS; ++k) s += W_rw[(size_t)d * cS + k];
    wbar[d] = s * (1.0f / cS);
  }
  if (threadIdx.x == 0) {
    float s = 0.f;
    #pragma unroll
    for (int k = 0; k < cS; ++k) s += b_rw[k];
    wbar[cD] = s * (1.0f / cS);
  }
}

// ---- h_row *= (h_row . wbar + bbar) ----------------------------------------
__global__ __launch_bounds__(256) void rwscale_kernel(
    float* __restrict__ h, const float* __restrict__ wbar)
{
  __shared__ float sm[256];
  int row = blockIdx.x;
  float* hr = h + (size_t)row * cD;
  float s = 0.f;
  for (int d = threadIdx.x; d < cD; d += 256) s += hr[d] * wbar[d];
  float dot = block_reduce_sum256(s, sm);
  float scale = dot + wbar[cD];
  for (int d = threadIdx.x; d < cD; d += 256) hr[d] *= scale;
}

// ---- final LayerNorm -------------------------------------------------------
__global__ __launch_bounds__(256) void layernorm_kernel(
    const float* __restrict__ h, const float* __restrict__ g,
    const float* __restrict__ bta, float* __restrict__ out)
{
  __shared__ float sm[256];
  int row = blockIdx.x;
  const float* hr = h + (size_t)row * cD;
  float s = 0.f, s2 = 0.f;
  for (int d = threadIdx.x; d < cD; d += 256) { float v = hr[d]; s += v; s2 += v * v; }
  float mu  = block_reduce_sum256(s,  sm) / (float)cD;
  float ex2 = block_reduce_sum256(s2, sm) / (float)cD;
  float inv = rsqrtf(ex2 - mu * mu + 1e-5f);
  for (int d = threadIdx.x; d < cD; d += 256)
    out[(size_t)row * cD + d] = (hr[d] - mu) * inv * g[d] + bta[d];
}

// ---- host-side orchestration ----------------------------------------------
extern "C" void kernel_launch(void* const* d_in, const int* in_sizes, int n_in,
                              void* d_out, int out_size, void* d_ws, size_t ws_size,
                              hipStream_t stream) {
  const int*   tok    = (const int*)  d_in[0];
  const float* mask   = (const float*)d_in[1];
  const float* emb    = (const float*)d_in[2];
  const float* norm_w = (const float*)d_in[3];
  const float* W_in   = (const float*)d_in[4];
  const float* conv_w = (const float*)d_in[5];
  const float* conv_b = (const float*)d_in[6];
  const float* W_x    = (const float*)d_in[7];
  const float* W_dt   = (const float*)d_in[8];
  const float* b_dt   = (const float*)d_in[9];
  const float* A_log  = (const float*)d_in[10];
  const float* D_skip = (const float*)d_in[11];
  const float* W_out  = (const float*)d_in[12];
  const float* W_rw   = (const float*)d_in[13];
  const float* b_rw   = (const float*)d_in[14];
  const float* ln_g   = (const float*)d_in[15];
  const float* ln_b   = (const float*)d_in[16];
  const float* head_W = (const float*)d_in[17];

  // workspace carve-up (~97 MiB)
  char* p = (char*)d_ws;
  float* hbuf     = (float*)p; p += sizeof(float) * (size_t)cM * cD;      // 8 MiB
  float* xnbuf    = (float*)p; p += sizeof(float) * (size_t)cM * cD;      // 8 MiB
  float* uzbuf    = (float*)p; p += sizeof(float) * (size_t)cM * 2 * cE;  // 32 MiB
  float* ubuf     = (float*)p; p += sizeof(float) * (size_t)cM * cE;      // 16 MiB
  float* dbcbuf   = (float*)p; p += sizeof(float) * (size_t)cM * cDBC;    // .75 MiB
  float* deltabuf = (float*)p; p += sizeof(float) * (size_t)cM * cE;      // 16 MiB
  float* ybuf     = (float*)p; p += sizeof(float) * (size_t)cM * cE;      // 16 MiB
  float* wbarbuf  = (float*)p; p += sizeof(float) * (size_t)(cD + 1);

  const int nElemD = cM * cD;   // 2,097,152
  const int nElemE = cM * cE;   // 4,194,304

  embed_kernel<<<(nElemD + 255) / 256, 256, 0, stream>>>(tok, mask, emb, hbuf);

  for (int i = 0; i < cNL; ++i) {
    const float* norm_w_l = norm_w + (size_t)i * cD;
    const float* W_in_l   = W_in   + (size_t)i * cD * 2 * cE;
    const float* conv_w_l = conv_w + (size_t)i * cE * cK;
    const float* conv_b_l = conv_b + (size_t)i * cE;
    const float* W_x_l    = W_x    + (size_t)i * cE * cDBC;
    const float* W_dt_l   = W_dt   + (size_t)i * cR * cE;
    const float* b_dt_l   = b_dt   + (size_t)i * cE;
    const float* A_log_l  = A_log  + (size_t)i * cE * cS;
    const float* D_skip_l = D_skip + (size_t)i * cE;
    const float* W_out_l  = W_out  + (size_t)i * cE * cD;
    const float* W_rw_l   = W_rw   + (size_t)i * cD * cS;
    const float* b_rw_l   = b_rw   + (size_t)i * cS;

    // xn = rmsnorm(h) * norm_w
    rmsnorm_kernel<<<cM, 256, 0, stream>>>(hbuf, norm_w_l, xnbuf);

    // uz = xn @ W_in   (2048x1024 @ 1024x4096)
    gemm_bf16_wmma<<<dim3(2 * cE / BN, cM / BM), 128, 0, stream>>>(
        xnbuf, cD, W_in_l, nullptr, uzbuf, cM, 2 * cE, cD);

    // u = silu(dwconv(u) + b)
    conv_silu_kernel<<<(nElemE + 255) / 256, 256, 0, stream>>>(
        uzbuf, conv_w_l, conv_b_l, ubuf);

    // dbc = u @ W_x    (2048x2048 @ 2048x96)
    gemm_bf16_wmma<<<dim3((cDBC + BN - 1) / BN, cM / BM), 128, 0, stream>>>(
        ubuf, cE, W_x_l, nullptr, dbcbuf, cM, cDBC, cE);

    // delta_pre = dr @ W_dt  (A = dbc cols 0..63, lda=96, Kd=64)
    gemm_bf16_wmma<<<dim3(cE / BN, cM / BM), 128, 0, stream>>>(
        dbcbuf, cDBC, W_dt_l, nullptr, deltabuf, cM, cE, cR);

    // delta = softplus(delta_pre + b_dt)
    softplus_kernel<<<(nElemE + 255) / 256, 256, 0, stream>>>(deltabuf, b_dt_l);

    // y = selective_scan(...) + u*D_skip
    scan_kernel<<<dim3(cE / 256, cB), 256, 0, stream>>>(
        ubuf, deltabuf, A_log_l, dbcbuf, D_skip_l, ybuf);

    // y *= silu(z)
    gate_kernel<<<(nElemE + 255) / 256, 256, 0, stream>>>(ybuf, uzbuf);

    // h = h + y @ W_out  (residual fused into GEMM epilogue, in-place safe)
    gemm_bf16_wmma<<<dim3(cD / BN, cM / BM), 128, 0, stream>>>(
        ybuf, cE, W_out_l, hbuf, hbuf, cM, cD, cE);

    // h *= mean(h @ W_rw + b_rw)  ==  h * (h . rowmean(W_rw) + mean(b_rw))
    wbar_kernel<<<1, 256, 0, stream>>>(W_rw_l, b_rw_l, wbarbuf);
    rwscale_kernel<<<cM, 256, 0, stream>>>(hbuf, wbarbuf);
  }

  // final LayerNorm into xnbuf, then LM head GEMM into d_out
  layernorm_kernel<<<cM, 256, 0, stream>>>(hbuf, ln_g, ln_b, xnbuf);
  gemm_bf16_wmma<<<dim3(cV / BN, cM / BM), 128, 0, stream>>>(
      xnbuf, cD, head_W, nullptr, (float*)d_out, cM, cV, cD);
}
